// SelfAttention_20418274525598
// MI455X (gfx1250) — compile-verified, matrix operand order checked by reading, not run on hardware
//
#include <hip/hip_runtime.h>
#include <hip/hip_bf16.h>

typedef __bf16 bf16_t;
typedef __attribute__((ext_vector_type(16))) __bf16 v16bf;
typedef __attribute__((ext_vector_type(8)))  __bf16 v8bf;
typedef __attribute__((ext_vector_type(8)))  float  v8f;

// ---------- bf16 <-> f32 via bit ops (no reliance on __bf16 arithmetic) ----------
__device__ __forceinline__ bf16_t f2bf(float f) {
    union { float f; unsigned u; } v; v.f = f;
    unsigned r = (v.u + 0x7FFFu + ((v.u >> 16) & 1u)) >> 16;
    union { unsigned short s; bf16_t b; } o; o.s = (unsigned short)r;
    return o.b;
}
__device__ __forceinline__ float bf2f(bf16_t b) {
    union { unsigned short s; bf16_t b; } i; i.b = b;
    union { unsigned u; float f; } o; o.u = ((unsigned)i.s) << 16;
    return o.f;
}

// ---------- WMMA helpers ----------
// Per-lane 16-bit A/B fragment (16x32 A or 32x16 B): lane holds row (A: M=lane&15)
// or column (B: N=lane&15); K indices = {8h..8h+7} U {16+8h..16+8h+7}, h=lane>>4.
// Both chunks are contiguous in memory when K is the contiguous axis -> two 16B loads.
__device__ __forceinline__ v16bf load_frag(const bf16_t* rowp, int half) {
    v8bf lo = *(const v8bf*)(rowp + 8 * half);
    v8bf hi = *(const v8bf*)(rowp + 16 + 8 * half);
    v16bf r;
#pragma unroll
    for (int j = 0; j < 8; ++j) { r[j] = lo[j]; r[j + 8] = hi[j]; }
    return r;
}

__device__ __forceinline__ v8f wmma_bf16(v16bf a, v16bf b, v8f c) {
    return __builtin_amdgcn_wmma_f32_16x16x32_bf16(false, a, false, b, (short)0, c, false, false);
}

// ================= K0: zero =================
__global__ void k_zero(float* p, int n) {
    int i = blockIdx.x * 256 + threadIdx.x;
    if (i < n) p[i] = 0.f;
}

// ================= K1: depthwise conv 3x3 stride 2 + stats =================
// block per (tsel, b, fea, c); 256 threads; 1024 output pixels
__global__ void k_dwconv(const float* __restrict__ x,
                         const float* __restrict__ wq, const float* __restrict__ wk,
                         const float* __restrict__ wv,
                         float* __restrict__ Yq, float* __restrict__ Yk, float* __restrict__ Yv,
                         float* __restrict__ sumA, float* __restrict__ sumQ) {
    int bid = blockIdx.x;
    int c = bid & 255;
    int fea = (bid >> 8) & 1;
    int b = (bid >> 9) & 1;
    int tsel = bid >> 10;
    const float* wsrc = (tsel == 0 ? wq : tsel == 1 ? wk : wv) + (size_t)(fea * 256 + c) * 9;
    float* Y = (tsel == 0 ? Yq : tsel == 1 ? Yk : Yv);
    float w9[9];
#pragma unroll
    for (int i = 0; i < 9; ++i) w9[i] = wsrc[i];
    const float* xin = x + ((size_t)b * 8192 + (size_t)fea * 4096) * 256 + c;
    float ls = 0.f, lq = 0.f;
    for (int pp = threadIdx.x; pp < 1024; pp += 256) {
        int y2 = pp >> 5, x2 = pp & 31;
        float acc = 0.f;
#pragma unroll
        for (int dy = 0; dy < 3; ++dy) {
            int yy = 2 * y2 - 1 + dy;
            if (yy < 0 || yy >= 64) continue;
#pragma unroll
            for (int dx = 0; dx < 3; ++dx) {
                int xx = 2 * x2 - 1 + dx;
                if (xx < 0 || xx >= 64) continue;
                acc += w9[dy * 3 + dx] * xin[(size_t)(yy * 64 + xx) * 256];
            }
        }
        size_t row = (size_t)b * 2048 + (size_t)fea * 1024 + pp;
        Y[row * 256 + c] = acc;
        ls += acc; lq += acc * acc;
    }
    __shared__ float sA[256], sB[256];
    sA[threadIdx.x] = ls; sB[threadIdx.x] = lq;
    __syncthreads();
    for (int s = 128; s > 0; s >>= 1) {
        if (threadIdx.x < s) { sA[threadIdx.x] += sA[threadIdx.x + s]; sB[threadIdx.x] += sB[threadIdx.x + s]; }
        __syncthreads();
    }
    if (threadIdx.x == 0) {
        int sidx = tsel * 512 + fea * 256 + c;
        atomicAdd(&sumA[sidx], sA[0]);   // exactly 2 addends (b=0,1): deterministic
        atomicAdd(&sumQ[sidx], sB[0]);
    }
}

// ================= K2: BN finalize -> per-channel scale/shift =================
__global__ void k_bnfin(const float* __restrict__ sumA, const float* __restrict__ sumQ,
                        const float* gq, const float* bq, const float* gk, const float* bk,
                        const float* gv, const float* bv,
                        float* __restrict__ sc, float* __restrict__ tc) {
    int i = blockIdx.x * 256 + threadIdx.x;
    if (i >= 1536) return;
    int tsel = i >> 9; int fc = i & 511;
    const float* g = tsel == 0 ? gq : tsel == 1 ? gk : gv;
    const float* bb = tsel == 0 ? bq : tsel == 1 ? bk : bv;
    float mean = sumA[i] * (1.f / 2048.f);
    float var = sumQ[i] * (1.f / 2048.f) - mean * mean;
    float s = g[fc] * rsqrtf(var + 1e-5f);
    sc[i] = s; tc[i] = bb[fc] - mean * s;
}

// ================= K2b: apply BN, cast to bf16 =================
__global__ void k_bnapply(const float* __restrict__ Y, const float* __restrict__ sc,
                          const float* __restrict__ tc, bf16_t* __restrict__ Ybf, int tsel) {
    size_t i = (size_t)blockIdx.x * 256 + threadIdx.x;
    if (i >= (size_t)4096 * 256) return;
    int c = (int)(i & 255);
    int row = (int)(i >> 8);
    int fea = (row >> 10) & 1;
    int sidx = tsel * 512 + fea * 256 + c;
    Ybf[i] = f2bf(Y[i] * sc[sidx] + tc[sidx]);
}

// ================= Kw: weights -> bf16 transposed (WT[n][k]) =================
__global__ void k_wt(const float* Wq, const float* Wk, const float* Wv, const float* Wo,
                     bf16_t* __restrict__ WT) {
    int i = blockIdx.x * 256 + threadIdx.x;
    if (i >= 4 * 65536) return;
    int mat = i >> 16; int r = i & 65535; int n = r >> 8; int k = r & 255;
    const float* W = mat == 0 ? Wq : mat == 1 ? Wk : mat == 2 ? Wv : Wo;
    WT[i] = f2bf(W[k * 256 + n]);
}

// ================= K3: projections (bf16 WMMA GEMM) =================
// wave per (tsel, mtile, ntile); Q,K row-major; V stored transposed VT[b][d][t]
__global__ void k_proj(const bf16_t* __restrict__ Yqb, const bf16_t* __restrict__ Ykb,
                       const bf16_t* __restrict__ Yvb, const bf16_t* __restrict__ WT,
                       bf16_t* __restrict__ Qbf, bf16_t* __restrict__ Kbf, bf16_t* __restrict__ VT) {
    int wid = blockIdx.x;
    int tsel = wid / (256 * 16);
    int rem = wid - tsel * 256 * 16;
    int mt = rem >> 4, nt = rem & 15;
    int lane = threadIdx.x, half = lane >> 4, li = lane & 15;
    const bf16_t* A = tsel == 0 ? Yqb : tsel == 1 ? Ykb : Yvb;
    const bf16_t* BT = WT + (size_t)tsel * 65536;
    const bf16_t* arow = A + (size_t)(mt * 16 + li) * 256;
    const bf16_t* brow = BT + (size_t)(nt * 16 + li) * 256;
    v8f acc = {};
#pragma unroll
    for (int kc = 0; kc < 8; ++kc)
        acc = wmma_bf16(load_frag(arow + kc * 32, half), load_frag(brow + kc * 32, half), acc);
#pragma unroll
    for (int j = 0; j < 8; ++j) {
        int m = mt * 16 + j + 8 * half;
        int n = nt * 16 + li;
        bf16_t v = f2bf(acc[j]);
        if (tsel == 0) Qbf[(size_t)m * 256 + n] = v;
        else if (tsel == 1) Kbf[(size_t)m * 256 + n] = v;
        else {
            int b = m >> 11, l = m & 2047;
            VT[((size_t)b * 256 + n) * 2048 + l] = v;
        }
    }
}

// ================= K4: fold fuse_w[:,8:16] into prior BEFORE upsample =================
__global__ void k_fprior(const float* __restrict__ prior, const float* __restrict__ fw,
                         bf16_t* __restrict__ FP) {
    size_t i = (size_t)blockIdx.x * 256 + threadIdx.x;
    if (i >= (size_t)2 * 8 * 512 * 2048) return;
    int t = (int)(i & 2047);
    size_t r3 = i >> 11;
    int r = (int)(r3 & 511);
    int o = (int)((r3 >> 9) & 7);
    int b = (int)(r3 >> 12);
    float acc = 0.f;
#pragma unroll
    for (int c = 0; c < 8; ++c)
        acc += fw[o * 16 + 8 + c] * prior[(((size_t)b * 8 + c) * 512 + r) * 2048 + t];
    FP[i] = f2bf(acc);
}

// ================= K5: fused score = fuse(QK heads) + bilinear(fused prior) =================
// wave per (b, l-tile16, t-tile16): 8 head WMMAs, register mix, 4-tap upsample.
__global__ void k_score(const bf16_t* __restrict__ Qbf, const bf16_t* __restrict__ Kbf,
                        const bf16_t* __restrict__ FP, const float* __restrict__ fw,
                        const float* __restrict__ fb, bf16_t* __restrict__ F) {
    int wid = blockIdx.x;
    int tt = wid & 127; int lt = (wid >> 7) & 127; int b = wid >> 14;
    int lane = threadIdx.x, half = lane >> 4, li = lane & 15;
    int l0 = lt * 16, t0 = tt * 16;
    const bf16_t* arow = Qbf + (size_t)b * 2048 * 256 + (size_t)(l0 + li) * 256;
    const bf16_t* brow = Kbf + (size_t)b * 2048 * 256 + (size_t)(t0 + li) * 256;
    v8f S[8];
#pragma unroll
    for (int c = 0; c < 8; ++c) {
        v8f z = {};
        S[c] = wmma_bf16(load_frag(arow + c * 32, half), load_frag(brow + c * 32, half), z);
    }
    // geometry of this l-tile: fixed fea, fixed y2, x2 = x0 + m
    int fea = l0 >> 10;
    int p0 = l0 & 1023;
    int y2 = p0 >> 5, x0 = p0 & 31;
    float cy = 0.5f * y2 - 0.25f;
    int yl = (int)floorf(cy); float fy = cy - (float)yl;
    int y0c = min(max(yl, 0), 15), y1c = min(max(yl + 1, 0), 15);
    int tg = t0 + li;
    const float scale = 0.0625f; // 256^-0.5
#pragma unroll
    for (int o = 0; o < 8; ++o) {
        const bf16_t* FPo = FP + (((size_t)b * 8 + o) * 512) * 2048;
        float fwo[8];
#pragma unroll
        for (int c = 0; c < 8; ++c) fwo[c] = fw[o * 16 + c] * scale;
        float fbo = fb[o];
#pragma unroll
        for (int j = 0; j < 8; ++j) {
            int m = j + 8 * half;
            int x2 = x0 + m;
            float cx = 0.5f * x2 - 0.25f;
            int xl = (int)floorf(cx); float fx = cx - (float)xl;
            int x0c = min(max(xl, 0), 15), x1c = min(max(xl + 1, 0), 15);
            int rb = fea * 256 + y0c * 16;
            int rt = fea * 256 + y1c * 16;
            float R = (1.f - fy) * ((1.f - fx) * bf2f(FPo[(size_t)(rb + x0c) * 2048 + tg]) +
                                    fx * bf2f(FPo[(size_t)(rb + x1c) * 2048 + tg])) +
                      fy * ((1.f - fx) * bf2f(FPo[(size_t)(rt + x0c) * 2048 + tg]) +
                            fx * bf2f(FPo[(size_t)(rt + x1c) * 2048 + tg]));
            float v = fbo + R;
#pragma unroll
            for (int c = 0; c < 8; ++c) v += fwo[c] * S[c][j];
            F[(((size_t)b * 8 + o) * 2048 + (size_t)(l0 + m)) * 2048 + t0 + li] = f2bf(v);
        }
    }
}

// ================= K6: row softmax (in-place, f32 math) =================
__global__ void k_softmax(bf16_t* __restrict__ F) {
    size_t row = blockIdx.x;
    bf16_t* p = F + row * 2048;
    float v[8];
    float mx = -1e30f;
#pragma unroll
    for (int i = 0; i < 8; ++i) { v[i] = bf2f(p[threadIdx.x + i * 256]); mx = fmaxf(mx, v[i]); }
    __shared__ float red[256];
    red[threadIdx.x] = mx; __syncthreads();
    for (int s = 128; s > 0; s >>= 1) {
        if (threadIdx.x < s) red[threadIdx.x] = fmaxf(red[threadIdx.x], red[threadIdx.x + s]);
        __syncthreads();
    }
    mx = red[0]; __syncthreads();
    float ls = 0.f;
#pragma unroll
    for (int i = 0; i < 8; ++i) { v[i] = __expf(v[i] - mx); ls += v[i]; }
    red[threadIdx.x] = ls; __syncthreads();
    for (int s = 128; s > 0; s >>= 1) {
        if (threadIdx.x < s) red[threadIdx.x] += red[threadIdx.x + s];
        __syncthreads();
    }
    float inv = 1.f / red[0];
#pragma unroll
    for (int i = 0; i < 8; ++i) p[threadIdx.x + i * 256] = f2bf(v[i] * inv);
}

// ================= K7: attn @ V (bf16 WMMA, K = 2048) =================
__global__ void k_attnv(const bf16_t* __restrict__ Attn, const bf16_t* __restrict__ VT,
                        bf16_t* __restrict__ Oat) {
    int wid = blockIdx.x;
    int nt = wid & 1; int lt = (wid >> 1) & 127; int h = (wid >> 8) & 7; int b = wid >> 11;
    int lane = threadIdx.x, half = lane >> 4, li = lane & 15;
    const bf16_t* arow = Attn + (((size_t)b * 8 + h) * 2048 + (size_t)(lt * 16 + li)) * 2048;
    const bf16_t* brow = VT + ((size_t)b * 256 + h * 32 + nt * 16 + li) * 2048;
    v8f acc = {};
#pragma unroll 4
    for (int kc = 0; kc < 64; ++kc)
        acc = wmma_bf16(load_frag(arow + kc * 32, half), load_frag(brow + kc * 32, half), acc);
#pragma unroll
    for (int j = 0; j < 8; ++j) {
        int l = lt * 16 + j + 8 * half;
        int n = h * 32 + nt * 16 + li;
        Oat[((size_t)b * 2048 + l) * 256 + n] = f2bf(acc[j]);
    }
}

// ================= K8: output projection + bias (f32 out) =================
__global__ void k_outproj(const bf16_t* __restrict__ Oat, const bf16_t* __restrict__ WoT,
                          const float* __restrict__ bo, float* __restrict__ out) {
    int wid = blockIdx.x;
    int nt = wid & 15; int mt = wid >> 4;
    int lane = threadIdx.x, half = lane >> 4, li = lane & 15;
    const bf16_t* arow = Oat + (size_t)(mt * 16 + li) * 256;
    const bf16_t* brow = WoT + (size_t)(nt * 16 + li) * 256;
    v8f acc = {};
#pragma unroll
    for (int kc = 0; kc < 8; ++kc)
        acc = wmma_bf16(load_frag(arow + kc * 32, half), load_frag(brow + kc * 32, half), acc);
#pragma unroll
    for (int j = 0; j < 8; ++j) {
        int m = mt * 16 + j + 8 * half;
        int n = nt * 16 + li;
        out[(size_t)m * 256 + n] = acc[j] + bo[n];
    }
}

// ================= launch =================
extern "C" void kernel_launch(void* const* d_in, const int* in_sizes, int n_in,
                              void* d_out, int out_size, void* d_ws, size_t ws_size,
                              hipStream_t stream) {
    (void)in_sizes; (void)n_in; (void)out_size; (void)ws_size;
    const float* x      = (const float*)d_in[0];
    const float* prior  = (const float*)d_in[1];
    const float* cq_w   = (const float*)d_in[2];
    const float* cq_g   = (const float*)d_in[3];
    const float* cq_b   = (const float*)d_in[4];
    const float* ck_w   = (const float*)d_in[5];
    const float* ck_g   = (const float*)d_in[6];
    const float* ck_b   = (const float*)d_in[7];
    const float* cv_w   = (const float*)d_in[8];
    const float* cv_g   = (const float*)d_in[9];
    const float* cv_b   = (const float*)d_in[10];
    const float* Wq     = (const float*)d_in[11];
    const float* Wk     = (const float*)d_in[12];
    const float* Wv     = (const float*)d_in[13];
    const float* Wo     = (const float*)d_in[14];
    const float* bo     = (const float*)d_in[15];
    const float* fuse_w = (const float*)d_in[16];
    const float* fuse_b = (const float*)d_in[17];

    char* base = (char*)d_ws;
    size_t off = 0;
    auto alloc = [&](size_t bytes) -> void* {
        off = (off + 255) & ~(size_t)255;
        void* p = base + off;
        off += bytes;
        return p;
    };
    float*  stats = (float*)alloc(3072 * sizeof(float));          // sumA[1536] + sumQ[1536]
    float*  sc    = (float*)alloc(1536 * sizeof(float));
    float*  tc    = (float*)alloc(1536 * sizeof(float));
    float*  Yq    = (float*)alloc((size_t)4096 * 256 * 4);
    float*  Yk    = (float*)alloc((size_t)4096 * 256 * 4);
    float*  Yv    = (float*)alloc((size_t)4096 * 256 * 4);
    bf16_t* Yqb   = (bf16_t*)alloc((size_t)4096 * 256 * 2);
    bf16_t* Ykb   = (bf16_t*)alloc((size_t)4096 * 256 * 2);
    bf16_t* Yvb   = (bf16_t*)alloc((size_t)4096 * 256 * 2);
    bf16_t* WT    = (bf16_t*)alloc((size_t)4 * 65536 * 2);
    bf16_t* Qbf   = (bf16_t*)alloc((size_t)4096 * 256 * 2);
    bf16_t* Kbf   = (bf16_t*)alloc((size_t)4096 * 256 * 2);
    bf16_t* VT    = (bf16_t*)alloc((size_t)4096 * 256 * 2);
    bf16_t* FP    = (bf16_t*)alloc((size_t)2 * 8 * 512 * 2048 * 2);
    bf16_t* F     = (bf16_t*)alloc((size_t)2 * 8 * 2048 * 2048 * 2);
    bf16_t* Oat   = (bf16_t*)alloc((size_t)4096 * 256 * 2);

    k_zero<<<12, 256, 0, stream>>>(stats, 3072);
    k_dwconv<<<3072, 256, 0, stream>>>(x, cq_w, ck_w, cv_w, Yq, Yk, Yv, stats, stats + 1536);
    k_bnfin<<<6, 256, 0, stream>>>(stats, stats + 1536, cq_g, cq_b, ck_g, ck_b, cv_g, cv_b, sc, tc);
    k_bnapply<<<4096, 256, 0, stream>>>(Yq, sc, tc, Yqb, 0);
    k_bnapply<<<4096, 256, 0, stream>>>(Yk, sc, tc, Ykb, 1);
    k_bnapply<<<4096, 256, 0, stream>>>(Yv, sc, tc, Yvb, 2);
    k_wt<<<1024, 256, 0, stream>>>(Wq, Wk, Wv, Wo, WT);
    k_proj<<<12288, 32, 0, stream>>>(Yqb, Ykb, Yvb, WT, Qbf, Kbf, VT);
    k_fprior<<<65536, 256, 0, stream>>>(prior, fuse_w, FP);
    k_score<<<32768, 32, 0, stream>>>(Qbf, Kbf, FP, fuse_w, fuse_b, F);
    k_softmax<<<32768, 256, 0, stream>>>(F);
    k_attnv<<<4096, 32, 0, stream>>>(F, VT, Oat);
    k_outproj<<<4096, 32, 0, stream>>>(Oat, WT + 3 * 65536, bo, (float*)d_out);
}
